// MultiHeadAttention_80977313399438
// MI455X (gfx1250) — compile-verified, matrix operand order checked by reading, not run on hardware
//
#include <hip/hip_runtime.h>

// ---------------------------------------------------------------------------
// MultiHeadAttention forward for MI455X (gfx1250), bf16 WMMA + flash-softmax.
//   B=4, S=2048, D=768, H=12, E=64
// - Every WMMA operand load / result store is a contiguous 16B/32B access.
// - GEMM K-loops ping-pong two register buffer sets (no rotation copies,
//   no spills) so s_wait_loadcnt covers loads issued a full step earlier.
// - Attention stages K/V chunks into LDS once per block (4 waves share the
//   stream), double-buffered one chunk ahead via async global->LDS copies
//   (guarded by __has_builtin; falls back to register-relay staging).
// ---------------------------------------------------------------------------

typedef __attribute__((ext_vector_type(16))) __bf16 v16bf;
typedef __attribute__((ext_vector_type(8)))  __bf16 v8bf;
typedef __attribute__((ext_vector_type(8)))  float  v8f;
typedef __attribute__((ext_vector_type(4)))  int    v4i;

#define BB 4
#define SS 2048
#define DD 768
#define HH 12
#define EE 64
#define HE (HH * EE)   // 768

#define LDKP 72   // LDS K-chunk row pitch (elements): 144B -> conflict-free reads
#define LDVP 40   // LDS V-chunk row pitch (elements): 80B  -> conflict-free reads

#if defined(__has_builtin)
#if __has_builtin(__builtin_amdgcn_global_load_async_to_lds_b128) && \
    __has_builtin(__builtin_amdgcn_s_wait_asynccnt)
#define USE_ASYNC_LDS 1
#endif
#endif
#ifndef USE_ASYNC_LDS
#define USE_ASYNC_LDS 0
#endif

#define AS1 __attribute__((address_space(1)))
#define AS3 __attribute__((address_space(3)))

static __device__ __forceinline__ v8f zero_v8f() {
  v8f z;
#pragma unroll
  for (int j = 0; j < 8; ++j) z[j] = 0.0f;
  return z;
}

// Load a 16x32 bf16 tile in WMMA A/B operand layout from memory where the
// "16" index (M for A, N for B) has stride `ld` elements and the "32" (K)
// index is contiguous.
//   lanes 0-15 : idx = lane,      K = {0..7, 16..23}
//   lanes 16-31: idx = lane - 16, K = {8..15, 24..31}
static __device__ __forceinline__ v16bf load_tile16(const __bf16* base, int ld) {
  const int lane = threadIdx.x & 31;
  const int half = lane >> 4;
  const int idx  = lane & 15;
  const __bf16* p = base + (size_t)idx * ld + half * 8;
  v8bf lo = *(const v8bf*)(p);        // K = 8*half + 0..7
  v8bf hi = *(const v8bf*)(p + 16);   // K = 8*half + 16..23
  v16bf r;
#pragma unroll
  for (int i = 0; i < 8; ++i) { r[i] = lo[i]; r[i + 8] = hi[i]; }
  return r;
}

static __device__ __forceinline__ v8f wmma_bf16(v16bf a, v16bf b, v8f c) {
  return __builtin_amdgcn_wmma_f32_16x16x32_bf16(
      /*neg_a=*/false, a, /*neg_b=*/false, b,
      /*c_mod=*/(short)0, c, /*reuse_a=*/false, /*reuse_b=*/false);
}

// 16B global -> LDS copy (async if available, else register relay)
static __device__ __forceinline__ void cp_g2l_16B(__bf16* dst, const __bf16* src) {
#if USE_ASYNC_LDS
  __builtin_amdgcn_global_load_async_to_lds_b128(
      (AS1 v4i*)(src), (AS3 v4i*)(dst), 0, 0);
#else
  *(v8bf*)dst = *(const v8bf*)src;
#endif
}

// ---------------------------------------------------------------------------
// Prep: fp32 -> bf16 conversions (+ weight transposes for contiguous loads)
// ---------------------------------------------------------------------------
__global__ void k_cvt_f32_bf16(const float* __restrict__ src,
                               __bf16* __restrict__ dst, int n) {
  int i = blockIdx.x * blockDim.x + threadIdx.x;
  const int stride = gridDim.x * blockDim.x;
  for (; i < n; i += stride) dst[i] = (__bf16)src[i];
}

// W[h][d][e] (f32) -> WT[h][e][d] (bf16)
__global__ void k_transpose_hde(const float* __restrict__ W,
                                __bf16* __restrict__ WT) {
  const int n = HH * DD * EE;
  int i = blockIdx.x * blockDim.x + threadIdx.x;
  const int stride = gridDim.x * blockDim.x;
  for (; i < n; i += stride) {
    int h = i / (DD * EE);
    int r = i - h * DD * EE;
    int d = r / EE;
    int e = r - d * EE;
    WT[((size_t)h * EE + e) * DD + d] = (__bf16)W[i];
  }
}

// Wp[he][d] (f32) -> WpT[d][he] (bf16)
__global__ void k_transpose_p(const float* __restrict__ Wp,
                              __bf16* __restrict__ WpT) {
  const int n = HE * DD;
  int i = blockIdx.x * blockDim.x + threadIdx.x;
  const int stride = gridDim.x * blockDim.x;
  for (; i < n; i += stride) {
    int he = i / DD;
    int d  = i - he * DD;
    WpT[(size_t)d * HE + he] = (__bf16)Wp[i];
  }
}

// ---------------------------------------------------------------------------
// Q/K projection, transposed orientation: D' = W_tile(m=e) x X^T(n=s).
// Lane owns one s; elements are contiguous e -> 16B bf16 stores to [s][e].
// Ping-pong K-loop (step 64 = 2 chunks, two buffer sets, no copies).
// ---------------------------------------------------------------------------
__global__ void k_qk_gemm(const __bf16* __restrict__ xb,   // [B*S][D]
                          const __bf16* __restrict__ WT3,  // [3][H][E][D]
                          const float* __restrict__ bq,
                          const float* __restrict__ bk,
                          __bf16* __restrict__ qout,       // [B][H][S][E]
                          __bf16* __restrict__ kout) {
  const int wave  = threadIdx.x >> 5;
  const int mtile = blockIdx.x * (blockDim.x >> 5) + wave;
  const int h = blockIdx.y;
  const int z = blockIdx.z;                      // 0 = q, 1 = k
  const int r0 = mtile * 16;                     // row into [B*S]
  const __bf16* WTh = WT3 + (((size_t)z * HH + h) * EE) * DD;
  const __bf16* Xb  = xb + (size_t)r0 * DD;
  const float* bias = (z == 0 ? bq : bk) + h * EE;

  v8f acc[4];
#pragma unroll
  for (int n = 0; n < 4; ++n) acc[n] = zero_v8f();

  v16bf bX0 = load_tile16(Xb, DD);
  v16bf aW0[4];
#pragma unroll
  for (int n = 0; n < 4; ++n) aW0[n] = load_tile16(WTh + (size_t)(n * 16) * DD, DD);

  for (int d0 = 0; d0 < DD; d0 += 64) {
    const int d1 = d0 + 32;
    v16bf bX1 = load_tile16(Xb + d1, DD);
    v16bf aW1[4];
#pragma unroll
    for (int n = 0; n < 4; ++n)
      aW1[n] = load_tile16(WTh + (size_t)(n * 16) * DD + d1, DD);
#pragma unroll
    for (int n = 0; n < 4; ++n) acc[n] = wmma_bf16(aW0[n], bX0, acc[n]);
    const int d2 = d0 + 64;
    if (d2 < DD) {
      bX0 = load_tile16(Xb + d2, DD);
#pragma unroll
      for (int n = 0; n < 4; ++n)
        aW0[n] = load_tile16(WTh + (size_t)(n * 16) * DD + d2, DD);
    }
#pragma unroll
    for (int n = 0; n < 4; ++n) acc[n] = wmma_bf16(aW1[n], bX1, acc[n]);
  }

  const int lane = threadIdx.x & 31;
  const int half = lane >> 4;
  const int b  = r0 >> 11;                       // / SS
  const int sl = r0 & (SS - 1);
  __bf16* outp = (z == 0 ? qout : kout) +
                 (((size_t)b * HH + h) * SS + sl + (lane & 15)) * EE;
#pragma unroll
  for (int n = 0; n < 4; ++n) {
    v8bf pk;
#pragma unroll
    for (int j = 0; j < 8; ++j)
      pk[j] = (__bf16)(acc[n][j] + bias[n * 16 + half * 8 + j]);
    *(v8bf*)(outp + n * 16 + half * 8) = pk;
  }
}

// ---------------------------------------------------------------------------
// V projection, normal orientation: D = X(m=s) x W(n=e), stored transposed as
// vT [b][h][e][s] (lane owns e; elements contiguous s -> 16B stores).
// ---------------------------------------------------------------------------
__global__ void k_v_gemm(const __bf16* __restrict__ xb,    // [B*S][D]
                         const __bf16* __restrict__ WTv,   // [H][E][D]
                         const float* __restrict__ bv,
                         __bf16* __restrict__ vT) {        // [B][H][E][S]
  const int wave  = threadIdx.x >> 5;
  const int mtile = blockIdx.x * (blockDim.x >> 5) + wave;
  const int h = blockIdx.y;
  const int r0 = mtile * 16;
  const __bf16* WTh = WTv + ((size_t)h * EE) * DD;
  const __bf16* Xb  = xb + (size_t)r0 * DD;
  const float* bias = bv + h * EE;

  v8f acc[4];
#pragma unroll
  for (int n = 0; n < 4; ++n) acc[n] = zero_v8f();

  v16bf aX0 = load_tile16(Xb, DD);
  v16bf bW0[4];
#pragma unroll
  for (int n = 0; n < 4; ++n) bW0[n] = load_tile16(WTh + (size_t)(n * 16) * DD, DD);

  for (int d0 = 0; d0 < DD; d0 += 64) {
    const int d1 = d0 + 32;
    v16bf aX1 = load_tile16(Xb + d1, DD);
    v16bf bW1[4];
#pragma unroll
    for (int n = 0; n < 4; ++n)
      bW1[n] = load_tile16(WTh + (size_t)(n * 16) * DD + d1, DD);
#pragma unroll
    for (int n = 0; n < 4; ++n) acc[n] = wmma_bf16(aX0, bW0[n], acc[n]);
    const int d2 = d0 + 64;
    if (d2 < DD) {
      aX0 = load_tile16(Xb + d2, DD);
#pragma unroll
      for (int n = 0; n < 4; ++n)
        bW0[n] = load_tile16(WTh + (size_t)(n * 16) * DD + d2, DD);
    }
#pragma unroll
    for (int n = 0; n < 4; ++n) acc[n] = wmma_bf16(aX1, bW1[n], acc[n]);
  }

  const int lane = threadIdx.x & 31;
  const int half = lane >> 4;
  const int b  = r0 >> 11;
  const int sl = r0 & (SS - 1);
  __bf16* outp = vT + (((size_t)b * HH + h) * EE) * SS;
#pragma unroll
  for (int n = 0; n < 4; ++n) {
    const int e = n * 16 + (lane & 15);
    const float bi = bias[e];
    v8bf pk;
#pragma unroll
    for (int j = 0; j < 8; ++j) pk[j] = (__bf16)(acc[n][j] + bi);  // s = sl+8h+j
    *(v8bf*)(outp + (size_t)e * SS + sl + half * 8) = pk;
  }
}

// ---------------------------------------------------------------------------
// Flash attention with LDS-staged K/V chunks shared by the 4 waves of a block.
// Each wave owns 16 queries; the block streams 64 chunks of 32 keys.
// Chunk staging is double-buffered one chunk ahead (async global->LDS when
// available). Scores computed transposed (lane = query) so softmax reductions
// are in-lane + one shfl_xor(16); score registers pack directly into the
// P^T B-operand; O accumulated transposed.
// ---------------------------------------------------------------------------
__global__ void k_attn(const __bf16* __restrict__ q,    // [B][H][S][E]
                       const __bf16* __restrict__ k,    // [B][H][S][E]
                       const __bf16* __restrict__ vT,   // [B][H][E][S]
                       __bf16* __restrict__ att) {      // [B*S][H*E]
  __shared__ __bf16 sK[2][32 * LDKP];   // [key][e], padded pitch
  __shared__ __bf16 sV[2][64 * LDVP];   // [e][key], padded pitch

  const int wave = threadIdx.x >> 5;
  const int qt = blockIdx.x * (blockDim.x >> 5) + wave;
  const int h = blockIdx.y;
  const int b = blockIdx.z;
  const int q0 = qt * 16;
  const size_t headoff = ((size_t)b * HH + h) * (size_t)SS * EE;
  const __bf16* qbase = q + headoff + (size_t)q0 * EE;
  const __bf16* kbase = k + headoff;
  const __bf16* vTb   = vT + headoff;   // [E][S] for this head

  // Q as B-operand (n=query, k=e), held for the whole stream.
  v16bf qb0 = load_tile16(qbase + 0, EE);
  v16bf qb1 = load_tile16(qbase + 32, EE);

  v8f o[4];
#pragma unroll
  for (int n = 0; n < 4; ++n) o[n] = zero_v8f();
  float mrun = -3.0e38f;
  float lrun = 0.0f;
  const float scale = 0.125f;  // 1/sqrt(E)

  // cooperative stage of one 32-key chunk: K 32x64 rows, V^T 64x32 rows
  const int t = threadIdx.x;             // 0..127
  const int skey = t >> 2, sko = (t & 3) * 8;   // K: 4 threads per key row
  const int sve  = t >> 1, svo = (t & 1) * 16;  // V: 2 threads per e row
  auto stage = [&](int key0, int par) {
    const __bf16* gk = kbase + (size_t)key0 * EE;
    __bf16* dK = &sK[par][0];
    cp_g2l_16B(dK + skey * LDKP + sko,      gk + (size_t)skey * EE + sko);
    cp_g2l_16B(dK + skey * LDKP + sko + 32, gk + (size_t)skey * EE + sko + 32);
    const __bf16* gv = vTb + key0;
    __bf16* dV = &sV[par][0];
    cp_g2l_16B(dV + sve * LDVP + svo,     gv + (size_t)sve * SS + svo);
    cp_g2l_16B(dV + sve * LDVP + svo + 8, gv + (size_t)sve * SS + svo + 8);
  };

  stage(0, 0);

  const int NCH = SS / 32;               // 64 chunks
  for (int c = 0; c < NCH; ++c) {
#if USE_ASYNC_LDS
    __builtin_amdgcn_s_wait_asynccnt(0); // own staged loads landed in LDS
#endif
    __syncthreads();                     // chunk c visible to all 4 waves
    if (c + 1 < NCH) stage((c + 1) * 32, (c + 1) & 1);
    if (c + 2 < NCH)                     // L2 prefetch two chunks out
      __builtin_prefetch(kbase + (size_t)(c + 2) * 32 * EE, 0, 3);

    const __bf16* lk = &sK[c & 1][0];
    const __bf16* lv = &sV[c & 1][0];

    // S'[key][query] for keys 0..15 (s0) and 16..31 (s1) of this chunk
    v8f s0 = zero_v8f();
    v8f s1 = zero_v8f();
    {
      v16bf kt;
      kt = load_tile16(lk, LDKP);                  s0 = wmma_bf16(kt, qb0, s0);
      kt = load_tile16(lk + 32, LDKP);             s0 = wmma_bf16(kt, qb1, s0);
      kt = load_tile16(lk + 16 * LDKP, LDKP);      s1 = wmma_bf16(kt, qb0, s1);
      kt = load_tile16(lk + 16 * LDKP + 32, LDKP); s1 = wmma_bf16(kt, qb1, s1);
    }

    // online softmax (lane pair l, l+16 holds the same query)
    float cmax = -3.0e38f;
#pragma unroll
    for (int j = 0; j < 8; ++j) {
      s0[j] *= scale;
      s1[j] *= scale;
      cmax = fmaxf(cmax, fmaxf(s0[j], s1[j]));
    }
    cmax = fmaxf(cmax, __shfl_xor(cmax, 16, 32));
    const float mnew = fmaxf(mrun, cmax);
    const float fac = __expf(mrun - mnew);
    float p0[8], p1[8];
    float csum = 0.0f;
#pragma unroll
    for (int j = 0; j < 8; ++j) {
      p0[j] = __expf(s0[j] - mnew);
      p1[j] = __expf(s1[j] - mnew);
      csum += p0[j] + p1[j];
    }
    csum += __shfl_xor(csum, 16, 32);
    lrun = lrun * fac + csum;
    mrun = mnew;

    // P^T as B-operand: lanes<16 need keys {0..7,16..23} = {p0,p1};
    // lanes>=16 need {8..15,24..31} = likewise. Falls out for free.
    v16bf pb;
#pragma unroll
    for (int j = 0; j < 8; ++j) {
      pb[j]     = (__bf16)p0[j];
      pb[j + 8] = (__bf16)p1[j];
    }

    // O' += V^T(chunk) x P^T
#pragma unroll
    for (int n = 0; n < 4; ++n) {
#pragma unroll
      for (int j = 0; j < 8; ++j) o[n][j] *= fac;
      v16bf vt = load_tile16(lv + (size_t)(n * 16) * LDVP, LDVP);
      o[n] = wmma_bf16(vt, pb, o[n]);
    }
  }

  // O'[e][q]: lane = query, elements = contiguous e -> 16B stores into
  // att[b*S + s][h*64 + e].
  const int lane = threadIdx.x & 31;
  const int half = lane >> 4;
  const float inv = 1.0f / lrun;
  const int row = b * SS + q0 + (lane & 15);
#pragma unroll
  for (int n = 0; n < 4; ++n) {
    v8bf pk;
#pragma unroll
    for (int j = 0; j < 8; ++j) pk[j] = (__bf16)(o[n][j] * inv);
    *(v8bf*)(att + (size_t)row * HE + h * EE + n * 16 + half * 8) = pk;
  }
}

// ---------------------------------------------------------------------------
// Output projection, transposed orientation: D' = Wp_tile(m=d) x att^T(n=s).
// Lane owns one s; elements are contiguous d -> 32B fp32 stores.
// ---------------------------------------------------------------------------
__global__ void k_proj(const __bf16* __restrict__ att,   // [B*S][HE]
                       const __bf16* __restrict__ WpT,   // [D][HE]
                       const float* __restrict__ bp,
                       float* __restrict__ out) {        // [B*S][D]
  const int wave  = threadIdx.x >> 5;
  const int mtile = blockIdx.x * (blockDim.x >> 5) + wave;
  const int d0 = blockIdx.y * 64;
  const int r0 = mtile * 16;
  const __bf16* Ab = att + (size_t)r0 * HE;
  const __bf16* Wb = WpT + (size_t)d0 * HE;

  v8f acc[4];
#pragma unroll
  for (int n = 0; n < 4; ++n) acc[n] = zero_v8f();

  v16bf bX0 = load_tile16(Ab, HE);
  v16bf aW0[4];
#pragma unroll
  for (int n = 0; n < 4; ++n) aW0[n] = load_tile16(Wb + (size_t)(n * 16) * HE, HE);

  for (int k0 = 0; k0 < HE; k0 += 64) {
    const int k1 = k0 + 32;
    v16bf bX1 = load_tile16(Ab + k1, HE);
    v16bf aW1[4];
#pragma unroll
    for (int n = 0; n < 4; ++n)
      aW1[n] = load_tile16(Wb + (size_t)(n * 16) * HE + k1, HE);
#pragma unroll
    for (int n = 0; n < 4; ++n) acc[n] = wmma_bf16(aW0[n], bX0, acc[n]);
    const int k2 = k0 + 64;
    if (k2 < HE) {
      bX0 = load_tile16(Ab + k2, HE);
#pragma unroll
      for (int n = 0; n < 4; ++n)
        aW0[n] = load_tile16(Wb + (size_t)(n * 16) * HE + k2, HE);
    }
#pragma unroll
    for (int n = 0; n < 4; ++n) acc[n] = wmma_bf16(aW1[n], bX1, acc[n]);
  }

  const int lane = threadIdx.x & 31;
  const int half = lane >> 4;
  const int row = r0 + (lane & 15);
#pragma unroll
  for (int n = 0; n < 4; ++n) {
    v8f r;
#pragma unroll
    for (int j = 0; j < 8; ++j)
      r[j] = acc[n][j] + bp[d0 + n * 16 + half * 8 + j];
    *(v8f*)(out + (size_t)row * DD + d0 + n * 16 + half * 8) = r;
  }
}

// ---------------------------------------------------------------------------
extern "C" void kernel_launch(void* const* d_in, const int* in_sizes, int n_in,
                              void* d_out, int out_size, void* d_ws,
                              size_t ws_size, hipStream_t stream) {
  (void)in_sizes; (void)n_in; (void)out_size; (void)ws_size;
  const float* x  = (const float*)d_in[0];
  const float* Wq = (const float*)d_in[1];
  const float* bq = (const float*)d_in[2];
  const float* Wk = (const float*)d_in[3];
  const float* bk = (const float*)d_in[4];
  const float* Wv = (const float*)d_in[5];
  const float* bv = (const float*)d_in[6];
  const float* Wp = (const float*)d_in[7];
  const float* bp = (const float*)d_in[8];
  float* out = (float*)d_out;

  char* ws = (char*)d_ws;
  size_t off = 0;
  auto walloc = [&](size_t bytes) -> char* {
    char* p = ws + off;
    off = (off + bytes + 255) & ~(size_t)255;
    return p;
  };
  __bf16* xb  = (__bf16*)walloc((size_t)BB * SS * DD * 2);       // x in bf16
  __bf16* WT3 = (__bf16*)walloc((size_t)3 * HH * EE * DD * 2);   // Wq/Wk/Wv^T
  __bf16* WpT = (__bf16*)walloc((size_t)DD * HE * 2);            // Wp^T
  __bf16* qb  = (__bf16*)walloc((size_t)BB * HH * SS * EE * 2);  // q [b,h,s,e]
  __bf16* kb  = (__bf16*)walloc((size_t)BB * HH * SS * EE * 2);  // k [b,h,s,e]
  __bf16* vT  = (__bf16*)walloc((size_t)BB * HH * EE * SS * 2);  // v^T [b,h,e,s]
  __bf16* att = (__bf16*)walloc((size_t)BB * SS * HE * 2);       // concat heads

  k_cvt_f32_bf16<<<2048, 256, 0, stream>>>(x, xb, BB * SS * DD);
  k_transpose_hde<<<1024, 256, 0, stream>>>(Wq, WT3 + (size_t)0 * HH * EE * DD);
  k_transpose_hde<<<1024, 256, 0, stream>>>(Wk, WT3 + (size_t)1 * HH * EE * DD);
  k_transpose_hde<<<1024, 256, 0, stream>>>(Wv, WT3 + (size_t)2 * HH * EE * DD);
  k_transpose_p<<<1024, 256, 0, stream>>>(Wp, WpT);

  // 512 M-tiles of 16 rows, 8 waves/block -> grid.x = 64
  k_qk_gemm<<<dim3(64, HH, 2), 256, 0, stream>>>(xb, WT3, bq, bk, qb, kb);
  k_v_gemm<<<dim3(64, HH), 256, 0, stream>>>(
      xb, WT3 + (size_t)2 * HH * EE * DD, bv, vT);

  // 128 query tiles of 16, 4 waves/block -> grid.x = 32
  k_attn<<<dim3(SS / 16 / 4, HH, BB), 128, 0, stream>>>(qb, kb, vT, att);

  // 512 M-tiles, 4 waves/block -> grid.x = 128; y = 12 column groups of 64
  k_proj<<<dim3(BB * SS / 16 / 4, DD / 64), 128, 0, stream>>>(att, WpT, bp, out);
}
